// MutualViewpointPositionalEncoding_32212254720759
// MI455X (gfx1250) — matrix operand order repaired and checked by reading, not compiled, not run on hardware
//
#include <hip/hip_runtime.h>
#include <hip/hip_bf16.h>
#include <math.h>

typedef __attribute__((ext_vector_type(16))) __bf16 v16bf;
typedef __attribute__((ext_vector_type(8)))  float  v8f;

#define HDIM 128
#define SBF 136   // bf16 row stride, K<=128 staging (272B = 17*16B)
#define SBM 264   // bf16 row stride, 256-wide double-duty buffer (528B = 33*16B)
#define SF  132   // f32 row stride

// ---- packed-weight workspace offsets (bf16 elements) ----
#define OFF_W2     ((size_t)0)
#define OFF_MDOWN  ((size_t)16384)
#define OFF_MUP    ((size_t)49152)
#define OFF_UW     ((size_t)57344)
#define OFF_FDOWN  ((size_t)73728)
#define OFF_FUP    ((size_t)81920)
#define OFF_OLDW   ((size_t)90112)
#define OFF_NEWW   ((size_t)106496)
// total 122880 bf16 = 245760 bytes of d_ws

union FragU { uint4 q[2]; v16bf v; };

// A-fragment (16x32 bf16) from LDS, per ISA A-matrix table:
// lanes 0-15: M=lane, K {0..7,16..23}; lanes 16-31: M=lane-16, K {8..15,24..31}
__device__ __forceinline__ v16bf load_a_frag(const __bf16* sA, int strideA, int kf, int lane) {
    int m  = lane & 15;
    int kh = lane >> 4;
    const __bf16* p = sA + m * strideA + (kf << 5) + (kh << 3);
    FragU f;
    f.q[0] = *(const uint4*)(p);
    f.q[1] = *(const uint4*)(p + 16);
    return f.v;
}

// B-fragment (32x16 bf16) pre-packed in global: frag = 512 bf16, lane owns 32 contiguous B
__device__ __forceinline__ v16bf load_b_frag(const __bf16* wbase, int fragidx, int lane) {
    const uint4* p = (const uint4*)(wbase + ((size_t)fragidx << 9)) + (lane << 1);
    FragU f;
    f.q[0] = p[0];
    f.q[1] = p[1];
    return f.v;
}

// 32xK @ Kx(NT*16): wave w owns N-tile w; B-frag reused across the two M-halves.
// kfBase: offset into the weight's k-fragment sequence (A column index stays local).
template<int KFR, int NT>
__device__ __forceinline__ void gemm2(const __bf16* sA, int strideA,
                                      const __bf16* wbase, int kfrTotal, int kfBase,
                                      int wave, int lane, v8f& c0, v8f& c1) {
    if (wave < NT) {
#pragma unroll
        for (int kf = 0; kf < KFR; ++kf) {
            v16bf b  = load_b_frag(wbase, wave * kfrTotal + kfBase + kf, lane);
            v16bf a0 = load_a_frag(sA,                strideA, kf, lane);
            v16bf a1 = load_a_frag(sA + 16 * strideA, strideA, kf, lane);
            c0 = __builtin_amdgcn_wmma_f32_16x16x32_bf16(false, a0, false, b, (short)0, c0, false, false);
            c1 = __builtin_amdgcn_wmma_f32_16x16x32_bf16(false, a1, false, b, (short)0, c1, false, false);
        }
    }
}

// Pack f32 weight [K][Nout] into bf16 B-fragments, frag order [ntile][kfrag];
// within frag: lanes 0-15 hold K {0..7,16..23} of col n, lanes 16-31 hold K {8..15,24..31} (ISA B layout).
__global__ void pack_weight(const float* __restrict__ W, __bf16* __restrict__ dst,
                            int K, int Nout) {
    int kfr   = K >> 5;
    int total = K * Nout;
    for (int i = blockIdx.x * blockDim.x + threadIdx.x; i < total;
         i += gridDim.x * blockDim.x) {
        int fragidx = i >> 9;
        int r    = i & 511;
        int lane = r >> 4, j = r & 15;
        int nt = fragidx / kfr, kf = fragidx - nt * kfr;
        int k  = (kf << 5) + ((lane >> 4) << 3) + (j & 7) + ((j >> 3) << 4);
        int n  = (nt << 4) + (lane & 15);
        dst[i] = (__bf16)W[k * Nout + n];
    }
}

__global__ __launch_bounds__(256)
void fused_edge_kernel(const float* __restrict__ pos_i, const float* __restrict__ pos_j,
                       const float* __restrict__ head_i, const float* __restrict__ head_j,
                       const long long* __restrict__ eidx, const float* __restrict__ edge_attr,
                       const float* __restrict__ nembs_i, const float* __restrict__ nembs_j,
                       const float* __restrict__ w1,  const float* __restrict__ b1,
                       const float* __restrict__ b2,  const float* __restrict__ mupb,
                       const float* __restrict__ ub,  const float* __restrict__ ung,
                       const float* __restrict__ unb, const float* __restrict__ fdb,
                       const float* __restrict__ fub, const float* __restrict__ eng,
                       const float* __restrict__ enb, const float* __restrict__ oldb,
                       const float* __restrict__ newb,
                       const __bf16* __restrict__ wpk,
                       float* __restrict__ out, int E)
{
    __shared__ __align__(16) __bf16 sX[32 * SBF];  // bf16 A staging (K<=128)
    __shared__ __align__(16) __bf16 sM[32 * SBM];  // 256-wide: [e_ij|e_ji] -> [n_i|n_j] -> [m_ij/t | edge_attr]
    __shared__ __align__(16) float  sE[32 * SF];   // e_ij f32 -> ffn f32 -> new_e f32
    __shared__ __align__(16) float  sU[32 * SF];   // u f32
    __shared__ float sRaw[2][32][8];
    __shared__ int   sSrc[32], sTgt[32];

    const int tid  = threadIdx.x;
    const int lane = tid & 31;
    const int wave = tid >> 5;
    const int e0   = blockIdx.x << 5;

    // D-layout constants (per ISA: VGPR r -> row r (+8 for lanes 16-31), col = lane&15)
    const int dn  = (wave << 4) + (lane & 15);   // this wave's output column
    const int dm0 = (lane >> 4) << 3;            // row base within a 16-row half

    // warm caches for the largest packed weight while geometry runs
    for (int off = tid * 128; off < 65536; off += 256 * 128)
        __builtin_prefetch((const char*)(wpk + OFF_MDOWN) + off, 0, 1);

    // ---- S0: geometry (atan2-free) ----
    if (tid < 32) {
        int e = e0 + tid; if (e >= E) e = E - 1;
        long long s = eidx[e], t = eidx[(size_t)E + e];
        sSrc[tid] = (int)s; sTgt[tid] = (int)t;
        float pix = pos_i[2 * s], piy = pos_i[2 * s + 1];
        float pjx = pos_j[2 * t], pjy = pos_j[2 * t + 1];
        float hi = head_i[s], hj = head_j[t];
        float dx = pix - pjx, dy = piy - pjy;
        float sj, cj; sincosf(hj, &sj, &cj);
        float si, ci; sincosf(hi, &si, &ci);
        float x  = dx * cj + dy * sj, y  = -dx * sj + dy * cj;
        float len = sqrtf(x * x + y * y);
        float inv = (len > 0.f) ? (1.f / len) : 0.f;
        float cd = ci * cj + si * sj;
        float sd = si * cj - ci * sj;
        sRaw[0][tid][0] = len;
        sRaw[0][tid][1] = (len > 0.f) ? x * inv : 1.f;
        sRaw[0][tid][2] = y * inv;
        sRaw[0][tid][3] = cd;
        sRaw[0][tid][4] = sd;
        float x2 = -dx * ci - dy * si, y2 = dx * si - dy * ci;
        sRaw[1][tid][0] = len;
        sRaw[1][tid][1] = (len > 0.f) ? x2 * inv : 1.f;
        sRaw[1][tid][2] = y2 * inv;
        sRaw[1][tid][3] = cd;
        sRaw[1][tid][4] = -sd;
    }
    __syncthreads();

    const int hrow = tid >> 3;           // 0..31  (row-layout passes: 8 threads/row, 16 cols each)
    const int hc0  = (tid & 7) << 4;

    // ---- S1: h1_ij = relu(raw@w1+b1) -> sX ----
    {
        float r0 = sRaw[0][hrow][0], r1 = sRaw[0][hrow][1], r2 = sRaw[0][hrow][2];
        float r3 = sRaw[0][hrow][3], r4 = sRaw[0][hrow][4];
#pragma unroll
        for (int j = 0; j < 16; ++j) {
            int c = hc0 + j;
            float a = b1[c] + r0 * w1[c] + r1 * w1[128 + c] + r2 * w1[256 + c]
                    + r3 * w1[384 + c] + r4 * w1[512 + c];
            sX[hrow * SBF + c] = (__bf16)fmaxf(a, 0.f);
        }
    }
    __syncthreads();

    // ---- S2: e_ij = h1_ij @ w2 ; +b2 -> f32 sE, bf16 sM[0..127] (in-register epilogue) ----
    {
        v8f c0 = {}, c1 = {};
        gemm2<4, 8>(sX, SBF, wpk + OFF_W2, 4, 0, wave, lane, c0, c1);
        float bv = b2[dn];
#pragma unroll
        for (int r = 0; r < 8; ++r) {
            float x0 = c0[r] + bv, x1 = c1[r] + bv;
            sE[(dm0 + r) * SF + dn]        = x0;
            sE[(16 + dm0 + r) * SF + dn]   = x1;
            sM[(dm0 + r) * SBM + dn]       = (__bf16)x0;
            sM[(16 + dm0 + r) * SBM + dn]  = (__bf16)x1;
        }
    }
    __syncthreads();

    // ---- S3: h1_ji -> sX ----
    {
        float r0 = sRaw[1][hrow][0], r1 = sRaw[1][hrow][1], r2 = sRaw[1][hrow][2];
        float r3 = sRaw[1][hrow][3], r4 = sRaw[1][hrow][4];
#pragma unroll
        for (int j = 0; j < 16; ++j) {
            int c = hc0 + j;
            float a = b1[c] + r0 * w1[c] + r1 * w1[128 + c] + r2 * w1[256 + c]
                    + r3 * w1[384 + c] + r4 * w1[512 + c];
            sX[hrow * SBF + c] = (__bf16)fmaxf(a, 0.f);
        }
    }
    __syncthreads();

    // ---- S4: e_ji = h1_ji @ w2 ; +b2 -> bf16 sM[128..255] ----
    {
        v8f c0 = {}, c1 = {};
        gemm2<4, 8>(sX, SBF, wpk + OFF_W2, 4, 0, wave, lane, c0, c1);
        float bv = b2[dn];
#pragma unroll
        for (int r = 0; r < 8; ++r) {
            sM[(dm0 + r) * SBM + 128 + dn]      = (__bf16)(c0[r] + bv);
            sM[(16 + dm0 + r) * SBM + 128 + dn] = (__bf16)(c1[r] + bv);
        }
    }
    __syncthreads();

    // ---- S5: m_mid = m_in(32x512) @ m_down, split K across two passes (acc in regs) ----
    {
        v8f mc0 = {}, mc1 = {};
        // pass A: K 0..255 = [e_ij | e_ji]
        gemm2<8, 4>(sM, SBM, wpk + OFF_MDOWN, 16, 0, wave, lane, mc0, mc1);
        __syncthreads();
        // overwrite buffer with gathered [n_i | n_j]
        for (int i = tid; i < 1024; i += 256) {
            int r = i >> 5, cc = (i & 31) << 2;
            float4 vi = *(const float4*)(nembs_i + (size_t)sSrc[r] * HDIM + cc);
            float4 vj = *(const float4*)(nembs_j + (size_t)sTgt[r] * HDIM + cc);
            __bf16* pi = sM + r * SBM + cc;
            __bf16* pj = sM + r * SBM + 128 + cc;
            pi[0] = (__bf16)vi.x; pi[1] = (__bf16)vi.y; pi[2] = (__bf16)vi.z; pi[3] = (__bf16)vi.w;
            pj[0] = (__bf16)vj.x; pj[1] = (__bf16)vj.y; pj[2] = (__bf16)vj.z; pj[3] = (__bf16)vj.w;
        }
        __syncthreads();
        // pass B: K 256..511 = [n_i | n_j]
        gemm2<8, 4>(sM, SBM, wpk + OFF_MDOWN, 16, 8, wave, lane, mc0, mc1);
        if (wave < 4) {
#pragma unroll
            for (int r = 0; r < 8; ++r) {
                sX[(dm0 + r) * SBF + dn]      = (__bf16)mc0[r];
                sX[(16 + dm0 + r) * SBF + dn] = (__bf16)mc1[r];
            }
        }
    }
    __syncthreads();

    // ---- S6: m_ij = relu(m_mid @ m_up + b) -> bf16 sM[0..127]; stage edge_attr -> sM[128..255] ----
    {
        v8f c0 = {}, c1 = {};
        gemm2<2, 8>(sX, SBF, wpk + OFF_MUP, 2, 0, wave, lane, c0, c1);
        float bv = mupb[dn];
#pragma unroll
        for (int r = 0; r < 8; ++r) {
            sM[(dm0 + r) * SBM + dn]      = (__bf16)fmaxf(c0[r] + bv, 0.f);
            sM[(16 + dm0 + r) * SBM + dn] = (__bf16)fmaxf(c1[r] + bv, 0.f);
        }
        for (int i = tid; i < 1024; i += 256) {
            int r = i >> 5, cc = (i & 31) << 2;
            int e = e0 + r; if (e >= E) e = E - 1;
            float4 v = *(const float4*)(edge_attr + (size_t)e * HDIM + cc);
            __bf16* p = sM + r * SBM + 128 + cc;
            p[0] = (__bf16)v.x; p[1] = (__bf16)v.y; p[2] = (__bf16)v.z; p[3] = (__bf16)v.w;
        }
    }
    __syncthreads();

    // ---- S7: u_pre = m_ij @ u_w ; +u_b -> f32 sU ----
    {
        v8f c0 = {}, c1 = {};
        gemm2<4, 8>(sM, SBM, wpk + OFF_UW, 4, 0, wave, lane, c0, c1);
        float bv = ub[dn];
#pragma unroll
        for (int r = 0; r < 8; ++r) {
            sU[(dm0 + r) * SF + dn]      = c0[r] + bv;
            sU[(16 + dm0 + r) * SF + dn] = c1[r] + bv;
        }
    }
    __syncthreads();

    // ---- S8: u = LN(u_pre + e_ij) -> sU f32, sX bf16 ----
    {
        float v[16], s = 0.f, s2 = 0.f;
#pragma unroll
        for (int j = 0; j < 16; ++j) {
            v[j] = sU[hrow * SF + hc0 + j] + sE[hrow * SF + hc0 + j];
            s += v[j]; s2 += v[j] * v[j];
        }
#pragma unroll
        for (int m = 1; m < 8; m <<= 1) { s += __shfl_xor(s, m, 8); s2 += __shfl_xor(s2, m, 8); }
        float mu  = s * (1.f / 128.f);
        float inv = rsqrtf(s2 * (1.f / 128.f) - mu * mu + 1e-5f);
#pragma unroll
        for (int j = 0; j < 16; ++j) {
            float y = (v[j] - mu) * inv * ung[hc0 + j] + unb[hc0 + j];
            sU[hrow * SF + hc0 + j]  = y;
            sX[hrow * SBF + hc0 + j] = (__bf16)y;
        }
    }
    __syncthreads();

    // ---- S9: t = relu(u @ ffn_down + b) -> bf16 sM[0..63] ----
    {
        v8f c0 = {}, c1 = {};
        gemm2<4, 4>(sX, SBF, wpk + OFF_FDOWN, 4, 0, wave, lane, c0, c1);
        if (wave < 4) {
            float bv = fdb[dn];
#pragma unroll
            for (int r = 0; r < 8; ++r) {
                sM[(dm0 + r) * SBM + dn]      = (__bf16)fmaxf(c0[r] + bv, 0.f);
                sM[(16 + dm0 + r) * SBM + dn] = (__bf16)fmaxf(c1[r] + bv, 0.f);
            }
        }
    }
    __syncthreads();

    // ---- S10: ffn = t @ ffn_up ; +b -> f32 sE (e_ij dead) ----
    {
        v8f c0 = {}, c1 = {};
        gemm2<2, 8>(sM, SBM, wpk + OFF_FUP, 2, 0, wave, lane, c0, c1);
        float bv = fub[dn];
#pragma unroll
        for (int r = 0; r < 8; ++r) {
            sE[(dm0 + r) * SF + dn]      = c0[r] + bv;
            sE[(16 + dm0 + r) * SF + dn] = c1[r] + bv;
        }
    }
    __syncthreads();

    // ---- S11: new_e = LN(ffn + u) -> sE f32, sX bf16 ----
    {
        float v[16], s = 0.f, s2 = 0.f;
#pragma unroll
        for (int j = 0; j < 16; ++j) {
            v[j] = sE[hrow * SF + hc0 + j] + sU[hrow * SF + hc0 + j];
            s += v[j]; s2 += v[j] * v[j];
        }
#pragma unroll
        for (int m = 1; m < 8; m <<= 1) { s += __shfl_xor(s, m, 8); s2 += __shfl_xor(s2, m, 8); }
        float mu  = s * (1.f / 128.f);
        float inv = rsqrtf(s2 * (1.f / 128.f) - mu * mu + 1e-5f);
#pragma unroll
        for (int j = 0; j < 16; ++j) {
            float y = (v[j] - mu) * inv * eng[hc0 + j] + enb[hc0 + j];
            sE[hrow * SF + hc0 + j]  = y;
            sX[hrow * SBF + hc0 + j] = (__bf16)y;
        }
    }
    __syncthreads();

    // ---- S12: old_proj & gate GEMMs stay in registers; fused sigmoid-blend -> global ----
    {
        v8f oc0 = {}, oc1 = {};          // edge_attr @ old_w
        gemm2<4, 8>(sM + 128, SBM, wpk + OFF_OLDW, 4, 0, wave, lane, oc0, oc1);
        v8f gc0 = {}, gc1 = {};          // new_e @ new_w
        gemm2<4, 8>(sX, SBF, wpk + OFF_NEWW, 4, 0, wave, lane, gc0, gc1);
        float ob = oldb[dn], nb = newb[dn];
#pragma unroll
        for (int r = 0; r < 8; ++r) {
            int row0 = dm0 + r, row1 = 16 + dm0 + r;
            float op0 = oc0[r] + ob, op1 = oc1[r] + ob;
            float pr0 = gc0[r] + nb, pr1 = gc1[r] + nb;
            float ne0 = sE[row0 * SF + dn], ne1 = sE[row1 * SF + dn];
            float g0 = 1.f / (1.f + __expf(-(op0 + pr0)));
            float g1 = 1.f / (1.f + __expf(-(op1 + pr1)));
            int eA = e0 + row0, eB = e0 + row1;
            if (eA < E) out[(size_t)eA * HDIM + dn] = ne0 + g0 * (op0 - ne0);
            if (eB < E) out[(size_t)eB * HDIM + dn] = ne1 + g1 * (op1 - ne1);
        }
    }
}

extern "C" void kernel_launch(void* const* d_in, const int* in_sizes, int n_in,
                              void* d_out, int out_size, void* d_ws, size_t ws_size,
                              hipStream_t stream) {
    const float*     pos_i  = (const float*)d_in[0];
    const float*     pos_j  = (const float*)d_in[1];
    const float*     head_i = (const float*)d_in[2];
    const float*     head_j = (const float*)d_in[3];
    const long long* eidx   = (const long long*)d_in[4];
    const float*     eattr  = (const float*)d_in[5];
    const float*     nei    = (const float*)d_in[6];
    const float*     nej    = (const float*)d_in[7];
    const float* w1   = (const float*)d_in[8];
    const float* b1   = (const float*)d_in[9];
    const float* w2   = (const float*)d_in[10];
    const float* b2   = (const float*)d_in[11];
    const float* mdn  = (const float*)d_in[12];
    const float* mup  = (const float*)d_in[13];
    const float* mupb = (const float*)d_in[14];
    const float* uw   = (const float*)d_in[15];
    const float* ubv  = (const float*)d_in[16];
    const float* ung  = (const float*)d_in[17];
    const float* unb  = (const float*)d_in[18];
    const float* fdw  = (const float*)d_in[19];
    const float* fdb  = (const float*)d_in[20];
    const float* fuw  = (const float*)d_in[21];
    const float* fub  = (const float*)d_in[22];
    const float* eng  = (const float*)d_in[23];
    const float* enb  = (const float*)d_in[24];
    const float* oldw = (const float*)d_in[25];
    const float* oldb = (const float*)d_in[26];
    const float* neww = (const float*)d_in[27];
    const float* newb = (const float*)d_in[28];

    int E = in_sizes[4] / 2;
    __bf16* wpk = (__bf16*)d_ws;

    pack_weight<<<(128 * 128 + 255) / 256, 256, 0, stream>>>(w2,   wpk + OFF_W2,    128, 128);
    pack_weight<<<(512 *  64 + 255) / 256, 256, 0, stream>>>(mdn,  wpk + OFF_MDOWN, 512,  64);
    pack_weight<<<( 64 * 128 + 255) / 256, 256, 0, stream>>>(mup,  wpk + OFF_MUP,    64, 128);
    pack_weight<<<(128 * 128 + 255) / 256, 256, 0, stream>>>(uw,   wpk + OFF_UW,    128, 128);
    pack_weight<<<(128 *  64 + 255) / 256, 256, 0, stream>>>(fdw,  wpk + OFF_FDOWN, 128,  64);
    pack_weight<<<( 64 * 128 + 255) / 256, 256, 0, stream>>>(fuw,  wpk + OFF_FUP,    64, 128);
    pack_weight<<<(128 * 128 + 255) / 256, 256, 0, stream>>>(oldw, wpk + OFF_OLDW,  128, 128);
    pack_weight<<<(128 * 128 + 255) / 256, 256, 0, stream>>>(neww, wpk + OFF_NEWW,  128, 128);

    int tiles = (E + 31) / 32;
    fused_edge_kernel<<<tiles, 256, 0, stream>>>(
        pos_i, pos_j, head_i, head_j, eidx, eattr, nei, nej,
        w1, b1, b2, mupb, ubv, ung, unb, fdb, fub, eng, enb, oldb, newb,
        wpk, (float*)d_out, E);
}
